// NNBeamformer_31379031065177
// MI455X (gfx1250) — matrix operand order repaired and checked by reading, not compiled, not run on hardware
//
#include <hip/hip_runtime.h>
#include <hip/hip_bf16.h>
#include <math.h>

#define N_EL   64
#define NTRI   2080          // 64*65/2
#define IN_DIM 4288          // 2*NTRI + 2*64
#define HID    512
#define BATCH  4096
#define O3PAD  192           // 129 padded to 3*64 for the GEMM tiler
#define O3REAL 129

typedef __attribute__((ext_vector_type(16))) __bf16 v16bf;
typedef __attribute__((ext_vector_type(8)))  __bf16 v8bf;
typedef __attribute__((ext_vector_type(8)))  float  v8f;

// ---------------------------------------------------------------------------
// Weight prep: W [In][Out] fp32 (torch-Linear transposed storage) ->
// WT [OutPad][In] bf16 (so WMMA B fragments are one contiguous 32B load/lane)
// ---------------------------------------------------------------------------
__global__ void k_wprep(const float* __restrict__ W, __bf16* __restrict__ WT,
                        int In, int Out, int OutPad) {
  int idx = blockIdx.x * blockDim.x + threadIdx.x;
  int total = OutPad * In;
  if (idx >= total) return;
  int n = idx / In;
  int k = idx - n * In;
  float v = (n < Out) ? W[(size_t)k * Out + n] : 0.0f;
  WT[(size_t)n * In + k] = (__bf16)v;
}

// ---------------------------------------------------------------------------
// Feature extraction: trace-normalized triu(R) + steering vector -> bf16
// feat[b] = [re(triu)/s (2080) | im(triu)/s (2080) | a.re (64) | a.im (64)]
// ---------------------------------------------------------------------------
__global__ void k_features(const double* __restrict__ R,   // [B][64][64] cplx interleaved
                           const double* __restrict__ an,  // [B][64] cplx interleaved
                           __bf16* __restrict__ feat) {    // [B][IN_DIM]
  __shared__ double red[64];
  __shared__ double s_inv;
  const int b = blockIdx.x, tid = threadIdx.x;
  const double* Rb = R + (size_t)b * (2 * N_EL * N_EL);
  if (tid < 64) red[tid] = Rb[(size_t)(tid * N_EL + tid) * 2];   // diag real
  __syncthreads();
  for (int s = 32; s > 0; s >>= 1) { if (tid < s) red[tid] += red[tid + s]; __syncthreads(); }
  if (tid == 0) s_inv = 1.0 / fmax(red[0] / 64.0, 1e-6);
  __syncthreads();
  const double inv = s_inv;
  __bf16* fb = feat + (size_t)b * IN_DIM;
  for (int t = tid; t < NTRI; t += blockDim.x) {
    // row-major triu enumeration: row i starts at S(i) = 64*i - i*(i-1)/2
    int i = (int)((129.0 - sqrt(129.0 * 129.0 - 8.0 * (double)t)) * 0.5);
    while (((i + 1) * 64 - ((i + 1) * i) / 2) <= t) ++i;
    while ((i * 64 - (i * (i - 1)) / 2) > t) --i;
    int j = i + (t - (i * 64 - (i * (i - 1)) / 2));
    const double* e = Rb + (size_t)(i * N_EL + j) * 2;
    fb[t]        = (__bf16)(float)(e[0] * inv);
    fb[NTRI + t] = (__bf16)(float)(e[1] * inv);
  }
  if (tid < 64) {
    const double* av = an + (size_t)b * (2 * N_EL) + tid * 2;
    fb[2 * NTRI + tid]      = (__bf16)(float)av[0];
    fb[2 * NTRI + 64 + tid] = (__bf16)(float)av[1];
  }
}

// ---------------------------------------------------------------------------
// bf16 WMMA GEMM: C[M,N] = A[M,K] x BT[N,K]^T + bias, optional ReLU.
// Per wave: 32x64 tile = 2 M-subtiles x 4 N-subtiles (8 accumulators);
// each B fragment is reused by 2 WMMAs, each A fragment by 4.
// Fragment layouts used here (contiguous k order, same for A and B):
//   A 16x32: lanes 0-15 hold K 0..15, lanes 16-31 hold K 16..31, contiguous per lane in [M][K]
//   B 32x16: lane half selects K 0..15 vs 16..31, contiguous per lane in [N][K]
//   C/D:     VGPR r -> row r (lanes 0-15) / r+8 (lanes 16-31), col = lane&15
// ---------------------------------------------------------------------------
__global__ void k_gemm_bf16(const __bf16* __restrict__ A, const __bf16* __restrict__ BT,
                            const float* __restrict__ bias, int nbias,
                            float* __restrict__ Cf, __bf16* __restrict__ Cb,
                            int M, int N, int K, int relu) {
  const int lane   = threadIdx.x & 31;
  const int wave   = threadIdx.x >> 5;
  const int nwaves = blockDim.x >> 5;
  const int m0 = (blockIdx.x * nwaves + wave) * 32;   // 2 M-subtiles per wave
  const int n0 = blockIdx.y * 64;                     // 4 N-subtiles per wave
  if (m0 >= M || n0 >= N) return;
  const int l16 = lane & 15;
  const int hi  = lane >> 4;
  const __bf16* Arow0 = A + (size_t)(m0 + l16) * K + hi * 16;
  const __bf16* Arow1 = Arow0 + (size_t)16 * K;
  const __bf16* Bp[4];
#pragma unroll
  for (int t = 0; t < 4; ++t)
    Bp[t] = BT + (size_t)(n0 + t * 16 + l16) * K + hi * 16;

  v8f acc[2][4] = {};
  for (int k0 = 0; k0 < K; k0 += 32) {
    if (k0 + 512 < K) {                       // stream-ahead on the A operand
      __builtin_prefetch(Arow0 + k0 + 512, 0, 1);
      __builtin_prefetch(Arow1 + k0 + 512, 0, 1);
    }
    v16bf af[2];
#pragma unroll
    for (int mt = 0; mt < 2; ++mt) {
      const __bf16* Ar = mt ? Arow1 : Arow0;
      v8bf a0 = *(const v8bf*)(Ar + k0);      // K = base+0..7
      v8bf a1 = *(const v8bf*)(Ar + k0 + 8);  // K = base+8..15
#pragma unroll
      for (int i = 0; i < 8; ++i) { af[mt][i] = a0[i]; af[mt][i + 8] = a1[i]; }
    }
#pragma unroll
    for (int t = 0; t < 4; ++t) {
      v16bf bfrag = *(const v16bf*)(Bp[t] + k0);
#pragma unroll
      for (int mt = 0; mt < 2; ++mt) {
        acc[mt][t] = __builtin_amdgcn_wmma_f32_16x16x32_bf16(
            false, af[mt], false, bfrag, (short)0, acc[mt][t], false, false);
      }
    }
  }
#pragma unroll
  for (int mt = 0; mt < 2; ++mt) {
#pragma unroll
    for (int t = 0; t < 4; ++t) {
      int n = n0 + t * 16 + l16;
      float bv = (n < nbias) ? bias[n] : 0.0f;
#pragma unroll
      for (int r = 0; r < 8; ++r) {
        int m = m0 + mt * 16 + r + hi * 8;
        float v = acc[mt][t][r] + bv;
        if (relu) v = fmaxf(v, 0.0f);
        if (Cf) Cf[(size_t)m * N + n] = v;
        if (Cb) Cb[(size_t)m * N + n] = (__bf16)v;
      }
    }
  }
}

// ---------------------------------------------------------------------------
// fp64 complex Cholesky solve, one 64-thread workgroup per batch item.
// Packed lower triangle in LDS (2080 * 16B = 33.3 KB << 320 KB/WGP).
// ---------------------------------------------------------------------------
__device__ __forceinline__ int TLO(int i) { return (i * (i + 1)) >> 1; }

__global__ void k_solve(const double* __restrict__ R, const double* __restrict__ an,
                        const float* __restrict__ out3,   // [B][O3PAD]
                        double* __restrict__ out) {       // [B][64] cplx interleaved
  __shared__ double2 Lp[NTRI];
  __shared__ double2 ac[64];
  __shared__ double2 xv[64];
  __shared__ double  red[64];
  __shared__ double  s_eps;
  const int b = blockIdx.x, tid = threadIdx.x;
  const double* Rb = R + (size_t)b * (2 * N_EL * N_EL);
  const float*  o3 = out3 + (size_t)b * O3PAD;

  // eps = (tr(R)/n) * exp(clip(out[0], -6, 2)) * 0.1
  red[tid] = Rb[(size_t)(tid * N_EL + tid) * 2];
  __syncthreads();
  for (int s = 32; s > 0; s >>= 1) { if (tid < s) red[tid] += red[tid + s]; __syncthreads(); }
  if (tid == 0) {
    double le = fmin(2.0, fmax(-6.0, (double)o3[0]));
    s_eps = (red[0] / 64.0) * exp(le) * 0.1;
  }
  __syncthreads();
  const double eps = s_eps;

  // a_corrected = a + 0.1*(out[1..64] + i*out[65..128])
  {
    const double* av = an + (size_t)b * (2 * N_EL) + tid * 2;
    double2 v;
    v.x = av[0] + 0.1 * (double)o3[1 + tid];
    v.y = av[1] + 0.1 * (double)o3[1 + 64 + tid];
    ac[tid] = v; xv[tid] = v;
  }
  // load lower triangle of R + eps*I (thread tid owns row tid)
  for (int j = 0; j <= tid; ++j) {
    const double* e = Rb + (size_t)(tid * N_EL + j) * 2;
    double2 v; v.x = e[0]; v.y = e[1];
    if (j == tid) { v.x += eps; v.y = 0.0; }
    Lp[TLO(tid) + j] = v;
  }
  __syncthreads();

  // right-looking complex Cholesky
  for (int k = 0; k < N_EL; ++k) {
    if (tid == 0) {
      double d = sqrt(fmax(Lp[TLO(k) + k].x, 1e-30));
      Lp[TLO(k) + k].x = d; Lp[TLO(k) + k].y = 0.0;
    }
    __syncthreads();
    double d = Lp[TLO(k) + k].x;
    if (tid > k) { Lp[TLO(tid) + k].x /= d; Lp[TLO(tid) + k].y /= d; }
    __syncthreads();
    if (tid > k) {
      double2 cik = Lp[TLO(tid) + k];
      for (int j = k + 1; j <= tid; ++j) {
        double2 ljk = Lp[TLO(j) + k];
        double2 a = Lp[TLO(tid) + j];
        a.x -= cik.x * ljk.x + cik.y * ljk.y;   // cik * conj(ljk)
        a.y -= cik.y * ljk.x - cik.x * ljk.y;
        Lp[TLO(tid) + j] = a;
      }
    }
    __syncthreads();
  }
  // forward: L y = a_corr
  for (int k = 0; k < N_EL; ++k) {
    if (tid == k) { double d = Lp[TLO(k) + k].x; xv[k].x /= d; xv[k].y /= d; }
    __syncthreads();
    if (tid > k) {
      double2 l = Lp[TLO(tid) + k], y = xv[k];
      xv[tid].x -= l.x * y.x - l.y * y.y;
      xv[tid].y -= l.x * y.y + l.y * y.x;
    }
    __syncthreads();
  }
  // backward: L^H x = y
  for (int k = N_EL - 1; k >= 0; --k) {
    if (tid == k) { double d = Lp[TLO(k) + k].x; xv[k].x /= d; xv[k].y /= d; }
    __syncthreads();
    if (tid < k) {
      double2 l = Lp[TLO(k) + tid], y = xv[k];   // conj(L(k,tid)) * y
      xv[tid].x -= l.x * y.x + l.y * y.y;
      xv[tid].y -= l.x * y.y - l.y * y.x;
    }
    __syncthreads();
  }
  // denom = Re(a_corr^H x); out = x / (denom + 1e-12)
  red[tid] = ac[tid].x * xv[tid].x + ac[tid].y * xv[tid].y;
  __syncthreads();
  for (int s = 32; s > 0; s >>= 1) { if (tid < s) red[tid] += red[tid + s]; __syncthreads(); }
  const double sc = 1.0 / (red[0] + 1e-12);
  double* ob = out + (size_t)b * (2 * N_EL);
  ob[tid * 2]     = xv[tid].x * sc;
  ob[tid * 2 + 1] = xv[tid].y * sc;
}

// ---------------------------------------------------------------------------
extern "C" void kernel_launch(void* const* d_in, const int* in_sizes, int n_in,
                              void* d_out, int out_size, void* d_ws, size_t ws_size,
                              hipStream_t stream) {
  (void)in_sizes; (void)n_in; (void)out_size; (void)ws_size;
  const double* R  = (const double*)d_in[0];
  const double* an = (const double*)d_in[1];
  const float*  W1 = (const float*)d_in[2];
  const float*  b1 = (const float*)d_in[3];
  const float*  W2 = (const float*)d_in[4];
  const float*  b2 = (const float*)d_in[5];
  const float*  W3 = (const float*)d_in[6];
  const float*  b3 = (const float*)d_in[7];

  char* ws = (char*)d_ws;
  size_t off = 0;
  auto alloc = [&](size_t bytes) -> void* {
    off = (off + 255) & ~(size_t)255;
    void* p = ws + off;
    off += bytes;
    return p;
  };
  __bf16* featbf = (__bf16*)alloc((size_t)BATCH * IN_DIM * 2);
  __bf16* w1t    = (__bf16*)alloc((size_t)HID * IN_DIM * 2);
  __bf16* w2t    = (__bf16*)alloc((size_t)HID * HID * 2);
  __bf16* w3t    = (__bf16*)alloc((size_t)O3PAD * HID * 2);
  __bf16* h1     = (__bf16*)alloc((size_t)BATCH * HID * 2);
  __bf16* h2     = (__bf16*)alloc((size_t)BATCH * HID * 2);
  float*  out3   = (float*)alloc((size_t)BATCH * O3PAD * 4);

  // weight prep (bf16 + transpose to [N][K])
  {
    int t1 = HID * IN_DIM, t2 = HID * HID, t3 = O3PAD * HID;
    k_wprep<<<(t1 + 255) / 256, 256, 0, stream>>>(W1, w1t, IN_DIM, HID, HID);
    k_wprep<<<(t2 + 255) / 256, 256, 0, stream>>>(W2, w2t, HID, HID, HID);
    k_wprep<<<(t3 + 255) / 256, 256, 0, stream>>>(W3, w3t, HID, O3REAL, O3PAD);
  }
  // features
  k_features<<<BATCH, 256, 0, stream>>>(R, an, featbf);

  // MLP: 8 waves/block, each wave = 32x64 tile -> block covers 256 rows x 64 cols
  dim3 blk(256);
  k_gemm_bf16<<<dim3(BATCH / 256, HID / 64), blk, 0, stream>>>(
      featbf, w1t, b1, HID, nullptr, h1, BATCH, HID, IN_DIM, 1);
  k_gemm_bf16<<<dim3(BATCH / 256, HID / 64), blk, 0, stream>>>(
      h1, w2t, b2, HID, nullptr, h2, BATCH, HID, HID, 1);
  k_gemm_bf16<<<dim3(BATCH / 256, O3PAD / 64), blk, 0, stream>>>(
      h2, w3t, b3, O3REAL, out3, nullptr, BATCH, O3PAD, HID, 0);

  // fp64 complex MVDR solve
  k_solve<<<BATCH, 64, 0, stream>>>(R, an, out3, (double*)d_out);
}
